// TextLSTM_30348238914178
// MI455X (gfx1250) — matrix operand order repaired and checked by reading, not compiled, not run on hardware
//
#include <hip/hip_runtime.h>
#include <math.h>

typedef __attribute__((ext_vector_type(16))) _Float16 v16h;
typedef __attribute__((ext_vector_type(8)))  _Float16 v8h;
typedef __attribute__((ext_vector_type(8)))  float    v8f;

#define B_SZ  512
#define T_SZ  128
#define DIN   128
#define H_SZ  256
#define G4H   1024        // 4*H
#define V_SZ  50257
#define V_PAD 51200       // 3200 n-tiles * 16 (padded, zero-filled)

__device__ __forceinline__ float sigf(float x) { return 1.0f / (1.0f + __expf(-x)); }

// ---------------- generic fp32 -> f16 weight convert -------------------------
__global__ void pack_w(const float* __restrict__ in, _Float16* __restrict__ out, long n) {
  for (long i = (long)blockIdx.x * blockDim.x + threadIdx.x; i < n;
       i += (long)gridDim.x * blockDim.x)
    out[i] = (_Float16)in[i];
}
// fcWh [V_PAD][256], zero-padded tail rows
__global__ void pack_fc(const float* __restrict__ W, _Float16* __restrict__ out) {
  const long n = (long)V_PAD * H_SZ;
  for (long i = (long)blockIdx.x * blockDim.x + threadIdx.x; i < n;
       i += (long)gridDim.x * blockDim.x) {
    long r = i / H_SZ, k = i - r * H_SZ;
    out[i] = (_Float16)((r < V_SZ) ? W[r * H_SZ + k] : 0.0f);
  }
}
__global__ void pack_bias(const float* __restrict__ a, const float* __restrict__ b,
                          float* __restrict__ o, int n) {
  for (int i = blockIdx.x * blockDim.x + threadIdx.x; i < n; i += gridDim.x * blockDim.x)
    o[i] = a[i] + b[i];
}

// ---------------- Xg = A @ Wih^T + bias over all (b,t): fully parallel -------
// A: [BT][K] (fp32 or f16), Wih: [1024][K] f16, out: [BT][1024] f16.
// One block = one 16-row M tile x all 1024 cols (8 waves x 8 n-tiles).
__global__ __launch_bounds__(256) void xg_gemm(
    const void* __restrict__ ain, int a_is_f32, int K,
    const _Float16* __restrict__ W, const float* __restrict__ bias,
    _Float16* __restrict__ out)
{
  __shared__ __attribute__((aligned(32))) _Float16 atile[16][256];
  const int tid = threadIdx.x;
  const int wave = tid >> 5, lane = tid & 31;
  const int lrow = lane & 15, lhalf = lane >> 4;
  const int m0 = blockIdx.x * 16;
  const int nkf = K >> 5;

  const float*    af = (const float*)ain;
  const _Float16* ah = (const _Float16*)ain;
  for (int i = tid; i < 16 * K; i += 256) {
    int r = i / K, k = i - r * K;
    size_t src = (size_t)(m0 + r) * K + k;
    atile[r][k] = a_is_f32 ? (_Float16)af[src] : ah[src];
  }
  __syncthreads();

  v8f acc[8];
  #pragma unroll
  for (int n = 0; n < 8; ++n) {
    float bv = bias[(wave * 8 + n) * 16 + lrow];
    #pragma unroll
    for (int r = 0; r < 8; ++r) acc[n][r] = bv;
  }
  for (int kf = 0; kf < nkf; ++kf) {
    const int kb = kf * 32;
    // batch the 8 B-fragment loads before the 8 WMMAs (partial waits, not 0)
    v16h bFs[8];
    #pragma unroll
    for (int n = 0; n < 8; ++n) {
      const int col = (wave * 8 + n) * 16 + lrow;
      bFs[n] = *(const v16h*)&W[(size_t)col * K + kb + 16 * lhalf];
    }
    v16h aF = *(const v16h*)&atile[lrow][kb + 16 * lhalf];
    #pragma unroll
    for (int n = 0; n < 8; ++n)
      acc[n] = __builtin_amdgcn_wmma_f32_16x16x32_f16(
          false, aF, false, bFs[n], (short)0, acc[n], false, false);
  }
  #pragma unroll
  for (int n = 0; n < 8; ++n) {
    const int col = (wave * 8 + n) * 16 + lrow;
    #pragma unroll
    for (int r = 0; r < 8; ++r)
      out[(size_t)(m0 + r + 8 * lhalf) * G4H + col] = (_Float16)acc[n][r];
  }
}

// ---------------- recurrent LSTM core: Whh register-resident -----------------
// One block owns 16 batch rows for all 128 steps. 512 threads = 16 waves;
// wave w owns gate cols [w*64, w*64+64) as 4 WMMA N-tiles. Whh B-fragments
// (4 tiles x 8 kf x 8 VGPR = 256 VGPRs) are loaded ONCE before the time loop:
// the per-step inner loop is pure WMMA fed from LDS A-fragments. Next step's
// Xg tile is prefetched (global_prefetch_b8) while the WMMA chain runs.
__global__ __launch_bounds__(512) void lstm_rec(
    const _Float16* __restrict__ Xg,    // [B][T][1024] f16 (bias folded in)
    const _Float16* __restrict__ Whh,   // [1024][256] f16
    _Float16* __restrict__ hout, int store_all)
{
  __shared__ __attribute__((aligned(32))) _Float16 abuf[16][H_SZ];   // h, f16
  __shared__ __attribute__((aligned(32))) _Float16 gates[16][G4H];   // h@Whh^T, f16

  const int tid = threadIdx.x;
  const int wave = tid >> 5, lane = tid & 31;
  const int lrow = lane & 15, lhalf = lane >> 4;
  const int b0 = blockIdx.x * 16;

  // register-resident Whh fragments for this wave's 4 N tiles
  v16h breg[4][8];
  #pragma unroll
  for (int n = 0; n < 4; ++n) {
    const int col = (wave * 4 + n) * 16 + lrow;
    #pragma unroll
    for (int kf = 0; kf < 8; ++kf)
      breg[n][kf] = *(const v16h*)&Whh[(size_t)col * H_SZ + kf * 32 + 16 * lhalf];
  }

  // activation-phase ownership: thread -> row = tid>>5, j = (tid&31) + 32*u
  const int drow = tid >> 5;
  const int jbase = tid & 31;
  float creg[8];
  #pragma unroll
  for (int u = 0; u < 8; ++u) creg[u] = 0.0f;

  for (int i = tid; i < 16 * H_SZ; i += 512)
    abuf[i >> 8][i & 255] = (_Float16)0.0f;      // h0 = 0
  __syncthreads();

  // warm the cache for the first Xg tile
  {
    const _Float16* pf = Xg + ((size_t)(b0 + drow) * T_SZ + 0) * G4H + jbase * 32;
    __builtin_prefetch((const void*)pf, 0, 0);
  }

  for (int t = 0; t < T_SZ; ++t) {
    // prefetch next step's Xg tile (32 KB spread over 512 threads) so the
    // HBM fetch overlaps this step's WMMA chain
    if (t + 1 < T_SZ) {
      const _Float16* pf = Xg + ((size_t)(b0 + drow) * T_SZ + (t + 1)) * G4H + jbase * 32;
      __builtin_prefetch((const void*)pf, 0, 0);
    }

    // Phase B: recurrent GEMM, pure WMMA (no global traffic)
    v8f acc[4];
    #pragma unroll
    for (int n = 0; n < 4; ++n)
      #pragma unroll
      for (int r = 0; r < 8; ++r) acc[n][r] = 0.0f;
    #pragma unroll
    for (int kf = 0; kf < 8; ++kf) {
      const int kb = kf * 32;
      v16h aF = *(const v16h*)&abuf[lrow][kb + 16 * lhalf];
      #pragma unroll
      for (int n = 0; n < 4; ++n)
        acc[n] = __builtin_amdgcn_wmma_f32_16x16x32_f16(
            false, aF, false, breg[n][kf], (short)0, acc[n], false, false);
    }
    // Phase C: spill to LDS for cross-wave gate math
    #pragma unroll
    for (int n = 0; n < 4; ++n) {
      const int col = (wave * 4 + n) * 16 + lrow;
      #pragma unroll
      for (int r = 0; r < 8; ++r) gates[r + 8 * lhalf][col] = (_Float16)acc[n][r];
    }
    __syncthreads();

    // Phase D: gates += Xg (coalesced, prefetched), activations, c/h update
    const bool wr = (store_all != 0) || (t == T_SZ - 1);
    const _Float16* xrow = Xg + ((size_t)(b0 + drow) * T_SZ + t) * G4H;
    #pragma unroll
    for (int u = 0; u < 8; ++u) {
      const int j = jbase + 32 * u;
      float ig = sigf((float)gates[drow][j]             + (float)xrow[j]);
      float fg = sigf((float)gates[drow][H_SZ + j]      + (float)xrow[H_SZ + j]);
      float gg = tanhf((float)gates[drow][2 * H_SZ + j] + (float)xrow[2 * H_SZ + j]);
      float og = sigf((float)gates[drow][3 * H_SZ + j]  + (float)xrow[3 * H_SZ + j]);
      float c  = fg * creg[u] + ig * gg;
      creg[u]  = c;
      float h  = og * tanhf(c);
      abuf[drow][j] = (_Float16)h;
      if (wr) {
        size_t dst = store_all ? (((size_t)(b0 + drow) * T_SZ + t) * H_SZ + j)
                               : ((size_t)(b0 + drow) * H_SZ + j);
        hout[dst] = (_Float16)h;
      }
    }
    __syncthreads();
  }
}

// ---------------- FC: logits[512 x V] = h_last @ fcW^T + fcb (WMMA) ----------
__global__ __launch_bounds__(256) void fc_logits(
    const _Float16* __restrict__ hlast,   // [512][256] f16
    const _Float16* __restrict__ Wfc,     // [V_PAD][256] f16
    const float* __restrict__ bfc,        // [V]
    float* __restrict__ out)              // [512][V]
{
  const int tid = threadIdx.x;
  const int wave = tid >> 5, lane = tid & 31;
  const int lrow = lane & 15, lhalf = lane >> 4;
  const int mtile = blockIdx.x;                      // 0..31
  const int nbase = blockIdx.y * 64 + wave * 8;      // n-tile base, < 3200

  const _Float16* arow = hlast + (size_t)(mtile * 16 + lrow) * H_SZ;
  v16h aF[8];
  #pragma unroll
  for (int kf = 0; kf < 8; ++kf) {
    aF[kf] = *(const v16h*)&arow[kf * 32 + 16 * lhalf];
  }
  #pragma unroll
  for (int n = 0; n < 8; ++n) {
    const int col = (nbase + n) * 16 + lrow;         // < 51200 (padded)
    float bv = (col < V_SZ) ? bfc[col] : 0.0f;
    v8f acc;
    #pragma unroll
    for (int r = 0; r < 8; ++r) acc[r] = bv;
    const _Float16* wrow = Wfc + (size_t)col * H_SZ + 16 * lhalf;
    // batch all 8 weight fragments (256B of the row per lane), then 8 WMMAs
    v16h bFs[8];
    #pragma unroll
    for (int kf = 0; kf < 8; ++kf) bFs[kf] = *(const v16h*)&wrow[kf * 32];
    #pragma unroll
    for (int kf = 0; kf < 8; ++kf)
      acc = __builtin_amdgcn_wmma_f32_16x16x32_f16(
          false, aF[kf], false, bFs[kf], (short)0, acc, false, false);
    if (col < V_SZ) {
      #pragma unroll
      for (int r = 0; r < 8; ++r) {
        int row = mtile * 16 + r + 8 * lhalf;
        out[(size_t)row * V_SZ + col] = acc[r];
      }
    }
  }
}

// ---------------- softmax over V per batch row (in place) --------------------
__global__ __launch_bounds__(256) void softmax_rows(float* __restrict__ out) {
  __shared__ float red[256];
  const int tid = threadIdx.x;
  float* p = out + (size_t)blockIdx.x * V_SZ;
  float m = -3.4e38f;
  for (int i = tid; i < V_SZ; i += 256) m = fmaxf(m, p[i]);
  red[tid] = m; __syncthreads();
  for (int s = 128; s > 0; s >>= 1) {
    if (tid < s) red[tid] = fmaxf(red[tid], red[tid + s]);
    __syncthreads();
  }
  m = red[0]; __syncthreads();
  float sum = 0.0f;
  for (int i = tid; i < V_SZ; i += 256) sum += __expf(p[i] - m);
  red[tid] = sum; __syncthreads();
  for (int s = 128; s > 0; s >>= 1) {
    if (tid < s) red[tid] += red[tid + s];
    __syncthreads();
  }
  float inv = 1.0f / red[0];
  for (int i = tid; i < V_SZ; i += 256) p[i] = __expf(p[i] - m) * inv;
}

// ---------------- launcher ---------------------------------------------------
extern "C" void kernel_launch(void* const* d_in, const int* in_sizes, int n_in,
                              void* d_out, int out_size, void* d_ws, size_t ws_size,
                              hipStream_t stream) {
  (void)in_sizes; (void)n_in; (void)out_size; (void)ws_size;
  const float* x    = (const float*)d_in[0];
  const float* Wih0 = (const float*)d_in[1];
  const float* Whh0 = (const float*)d_in[2];
  const float* bih0 = (const float*)d_in[3];
  const float* bhh0 = (const float*)d_in[4];
  const float* Wih1 = (const float*)d_in[5];
  const float* Whh1 = (const float*)d_in[6];
  const float* bih1 = (const float*)d_in[7];
  const float* bhh1 = (const float*)d_in[8];
  const float* fcW  = (const float*)d_in[9];
  const float* fcb  = (const float*)d_in[10];
  float* out = (float*)d_out;
  char* ws = (char*)d_ws;

  // workspace layout (256B-aligned offsets)
  _Float16* Wih0h = (_Float16*)(ws + 0);          // 1024*128*2 =   262144
  _Float16* Whh0h = (_Float16*)(ws + 262144);     // 1024*256*2 =   524288
  _Float16* Wih1h = (_Float16*)(ws + 786432);     // 1024*256*2 =   524288
  _Float16* Whh1h = (_Float16*)(ws + 1310720);    // 1024*256*2 =   524288
  _Float16* fcWh  = (_Float16*)(ws + 1835008);    // 51200*256*2 = 26214400
  float*    bias0 = (float*)   (ws + 28049408);   // 1024*4
  float*    bias1 = (float*)   (ws + 28053504);   // 1024*4
  _Float16* H0    = (_Float16*)(ws + 28057600);   // 512*128*256*2 = 33554432
  _Float16* H1l   = (_Float16*)(ws + 44834816);   // 512*256*2     =   262144
  _Float16* Xg    = (_Float16*)(ws + 45096960);   // 512*128*1024*2 = 134217728

  pack_w<<<128, 256, 0, stream>>>(Wih0, Wih0h, (long)G4H * DIN);
  pack_w<<<256, 256, 0, stream>>>(Whh0, Whh0h, (long)G4H * H_SZ);
  pack_w<<<256, 256, 0, stream>>>(Wih1, Wih1h, (long)G4H * H_SZ);
  pack_w<<<256, 256, 0, stream>>>(Whh1, Whh1h, (long)G4H * H_SZ);
  pack_fc<<<1024, 256, 0, stream>>>(fcW, fcWh);
  pack_bias<<<4, 256, 0, stream>>>(bih0, bhh0, bias0, G4H);
  pack_bias<<<4, 256, 0, stream>>>(bih1, bhh1, bias1, G4H);

  // layer 0: parallel input projection over all (b,t), then recurrence
  xg_gemm<<<B_SZ * T_SZ / 16, 256, 0, stream>>>((const void*)x, 1, DIN, Wih0h, bias0, Xg);
  lstm_rec<<<32, 512, 0, stream>>>(Xg, Whh0h, H0, 1);
  // layer 1
  xg_gemm<<<B_SZ * T_SZ / 16, 256, 0, stream>>>((const void*)H0, 0, H_SZ, Wih1h, bias1, Xg);
  lstm_rec<<<32, 512, 0, stream>>>(Xg, Whh1h, H1l, 0);

  fc_logits<<<dim3(32, 50), 256, 0, stream>>>(H1l, fcWh, fcb, out);
  softmax_rows<<<512, 256, 0, stream>>>(out);
}
